// MambaModule_747324310169
// MI455X (gfx1250) — matrix operand order repaired and checked by reading, not compiled, not run on hardware
//
#include <hip/hip_runtime.h>
#include <hip/hip_bf16.h>

// ---------------------------------------------------------------------------
// Types for CDNA5 WMMA
// ---------------------------------------------------------------------------
typedef __attribute__((ext_vector_type(16))) __bf16 v16bf;
typedef __attribute__((ext_vector_type(8)))  float  v8f;
typedef __attribute__((ext_vector_type(4)))  int    v4i;

union FragBF {
    v16bf v;
    uint4 q[2];
};

// ---------------------------------------------------------------------------
// CDNA5 async global->LDS path (ASYNCcnt-tracked), with safe fallback
// ---------------------------------------------------------------------------
#if __has_builtin(__builtin_amdgcn_global_load_async_to_lds_b128)
#define ASYNC_OK 1
#endif

__device__ __forceinline__ void cp16_async(void* ldsDst, const void* gSrc) {
#ifdef ASYNC_OK
    __builtin_amdgcn_global_load_async_to_lds_b128(
        (__attribute__((address_space(1))) v4i*)gSrc,
        (__attribute__((address_space(3))) v4i*)ldsDst,
        /*imm offset*/ 0, /*cpol*/ 0);
#else
    *(uint4*)ldsDst = *(const uint4*)gSrc;
#endif
}

#if __has_builtin(__builtin_amdgcn_s_wait_asynccnt)
#define WAIT_ASYNC(n) __builtin_amdgcn_s_wait_asynccnt(n)
#elif defined(ASYNC_OK)
#define WAIT_ASYNC(n) asm volatile("s_wait_asynccnt %0" ::"i"(n) : "memory")
#else
#define WAIT_ASYNC(n)
#endif

__device__ __forceinline__ unsigned short f2bf(float f) {
    unsigned u = __float_as_uint(f);
    unsigned r = (u + 0x7fffu + ((u >> 16) & 1u)) >> 16;   // RNE
    return (unsigned short)r;
}

__device__ __forceinline__ float siluf(float x) {
    return x / (1.0f + __expf(-x));
}
__device__ __forceinline__ float softplusf(float x) {
    return (x > 20.0f) ? x : log1pf(__expf(x));
}

// ---------------------------------------------------------------------------
// f32 -> bf16 conversion
// ---------------------------------------------------------------------------
__global__ __launch_bounds__(256) void cvt_bf16_k(const float* __restrict__ s,
                                                  unsigned short* __restrict__ d,
                                                  int n) {
    int i = blockIdx.x * 256 + threadIdx.x;
    if (i < n) d[i] = f2bf(s[i]);
}

// Concatenate Wb|Wc -> [D,32] bf16 and bb|bc -> [32] f32
__global__ __launch_bounds__(256) void concat_bc_k(const float* __restrict__ Wb,
                                                   const float* __restrict__ Wc,
                                                   const float* __restrict__ bb,
                                                   const float* __restrict__ bc,
                                                   unsigned short* __restrict__ Wbc,
                                                   float* __restrict__ biasBC,
                                                   int Dd) {
    int i = blockIdx.x * 256 + threadIdx.x;
    int total = Dd * 32;
    if (i < total) {
        int d = i >> 5, c = i & 31;
        float v = (c < 16) ? Wb[d * 16 + c] : Wc[d * 16 + (c - 16)];
        Wbc[i] = f2bf(v);
    }
    if (i < 32) biasBC[i] = (i < 16) ? bb[i] : bc[i - 16];
}

// ---------------------------------------------------------------------------
// Tiled bf16 WMMA GEMM: C[M,N] = act(A[M,K] @ B[K,N] + bias)
// BM=128 BN=64 BK=64, 256 threads = 8 waves, wave tile 32x32 (2x2 WMMA tiles)
// Double-buffered LDS fed by async global->LDS DMA (ASYNCcnt).
// EPI: 0=none 1=silu 2=softplus
// ---------------------------------------------------------------------------
#define BM 128
#define BN 64
#define BK 64
#define APAD 8
#define LDA (BK + APAD)   // 72 elems -> 144B row stride (bank-spread)
#define LDB (BN + APAD)   // 72

template <int EPI>
__global__ __launch_bounds__(256) void gemm_bf16_k(
    const unsigned short* __restrict__ A,
    const unsigned short* __restrict__ Bm,
    const float* __restrict__ bias,
    float* __restrict__ C,
    int M, int N, int K) {
    __shared__ __align__(16) unsigned short As[2][BM * LDA];
    __shared__ __align__(16) unsigned short Bs[2][BK * LDB];

    const int tid  = threadIdx.x;
    const int wave = tid >> 5;
    const int lane = tid & 31;
    const int wm   = wave >> 1;       // 0..3
    const int wn   = wave & 1;        // 0..1
    const int m0   = blockIdx.y * BM;
    const int n0   = blockIdx.x * BN;
    const int steps = K / BK;

    // 6 async 16B loads per thread per tile (A: 4, B: 2)
    auto issue_tile = [&](int step, int buf) {
        const int k0 = step * BK;
#pragma unroll
        for (int j = 0; j < 4; ++j) {           // A tile: 128x64 = 1024 chunks
            int ch = tid + j * 256;
            int r = ch >> 3, c = (ch & 7) * 8;
            cp16_async(&As[buf][r * LDA + c],
                       &A[(size_t)(m0 + r) * K + k0 + c]);
        }
#pragma unroll
        for (int j = 0; j < 2; ++j) {           // B tile: 64x64 = 512 chunks
            int ch = tid + j * 256;
            int r = ch >> 3, c = (ch & 7) * 8;
            cp16_async(&Bs[buf][r * LDB + c],
                       &Bm[(size_t)(k0 + r) * N + n0 + c]);
        }
    };

    v8f acc[2][2] = {};
    issue_tile(0, 0);
    if (steps > 1) issue_tile(1, 1);

    for (int i = 0; i < steps; ++i) {
        if (i + 1 < steps) { WAIT_ASYNC(6); }   // allow next tile in flight
        else              { WAIT_ASYNC(0); }
        __syncthreads();                        // tile i visible to all waves

        const int buf = i & 1;
        const int kh  = (lane >> 4) * 8;
#pragma unroll
        for (int kk = 0; kk < BK; kk += 32) {
            FragBF fa[2], fb[2];
            // A frag (16x32): lane<16 row=lane K{0..7,16..23};
            //                 lane>=16 row=lane-16 K{8..15,24..31}
#pragma unroll
            for (int mt = 0; mt < 2; ++mt) {
                int row = wm * 32 + mt * 16 + (lane & 15);
                fa[mt].q[0] = *(const uint4*)&As[buf][row * LDA + kk + kh];
                fa[mt].q[1] = *(const uint4*)&As[buf][row * LDA + kk + kh + 16];
            }
            // B frag (32x16): lane holds col N=lane&15, K same order as A frag
#pragma unroll
            for (int nt = 0; nt < 2; ++nt) {
                int col = wn * 32 + nt * 16 + (lane & 15);
#pragma unroll
                for (int s = 0; s < 16; ++s)
                    ((unsigned short*)&fb[nt])[s] =
                        Bs[buf][(kk + kh + (s < 8 ? s : s + 8)) * LDB + col];
            }
#pragma unroll
            for (int mt = 0; mt < 2; ++mt)
#pragma unroll
                for (int nt = 0; nt < 2; ++nt)
                    acc[mt][nt] = __builtin_amdgcn_wmma_f32_16x16x32_bf16(
                        false, fa[mt].v, false, fb[nt].v,
                        (short)0, acc[mt][nt], false, false);
        }
        __syncthreads();                        // all waves done reading buf
        if (i + 2 < steps) issue_tile(i + 2, buf);
    }

    // Epilogue: VGPR r holds M=r (lanes 0-15) / M=8+r (lanes 16-31), N=lane&15
#pragma unroll
    for (int mt = 0; mt < 2; ++mt)
#pragma unroll
        for (int nt = 0; nt < 2; ++nt) {
            int n = n0 + wn * 32 + nt * 16 + (lane & 15);
            float bv = bias[n];
#pragma unroll
            for (int r = 0; r < 8; ++r) {
                int m = m0 + wm * 32 + mt * 16 + r + ((lane >> 4) << 3);
                float v = acc[mt][nt][r] + bv;
                if (EPI == 1) v = siluf(v);
                else if (EPI == 2) v = softplusf(v);
                C[(size_t)m * N + n] = v;
            }
        }
}

// ---------------------------------------------------------------------------
// Narrow GEMM for B/C projection: C[M,32] = A[M,K] @ W[K,32] + bias
// ---------------------------------------------------------------------------
__global__ __launch_bounds__(256) void gemm_n32_k(
    const unsigned short* __restrict__ A,
    const unsigned short* __restrict__ W,
    const float* __restrict__ bias,
    float* __restrict__ C,
    int M, int K) {
    const int wave = threadIdx.x >> 5;
    const int lane = threadIdx.x & 31;
    const int m0   = blockIdx.x * 128 + wave * 16;
    const int row  = m0 + (lane & 15);
    const int kh   = (lane >> 4) * 8;

    v8f acc0 = {}, acc1 = {};
    for (int k0 = 0; k0 < K; k0 += 32) {
        FragBF fa, fb0, fb1;
        fa.q[0] = *(const uint4*)&A[(size_t)row * K + k0 + kh];
        fa.q[1] = *(const uint4*)&A[(size_t)row * K + k0 + kh + 16];
#pragma unroll
        for (int s = 0; s < 16; ++s) {
            int kr = k0 + kh + (s < 8 ? s : s + 8);
            ((unsigned short*)&fb0)[s] = W[(size_t)kr * 32 + (lane & 15)];
            ((unsigned short*)&fb1)[s] = W[(size_t)kr * 32 + 16 + (lane & 15)];
        }
        acc0 = __builtin_amdgcn_wmma_f32_16x16x32_bf16(
            false, fa.v, false, fb0.v, (short)0, acc0, false, false);
        acc1 = __builtin_amdgcn_wmma_f32_16x16x32_bf16(
            false, fa.v, false, fb1.v, (short)0, acc1, false, false);
    }
    int n = lane & 15;
#pragma unroll
    for (int r = 0; r < 8; ++r) {
        int m = m0 + r + ((lane >> 4) << 3);
        C[(size_t)m * 32 + n]      = acc0[r] + bias[n];
        C[(size_t)m * 32 + 16 + n] = acc1[r] + bias[16 + n];
    }
}

// ---------------------------------------------------------------------------
// Depthwise conv3 (same padding) + SiLU; emits f32 (scan) and bf16 (GEMMs)
// Shapes hardcoded: B=4, T=1024, D=1024
// ---------------------------------------------------------------------------
__global__ __launch_bounds__(256) void conv_silu_k(
    const float* __restrict__ x, const float* __restrict__ w,
    const float* __restrict__ cb,
    float* __restrict__ xcf, unsigned short* __restrict__ xcb, int total) {
    int i = blockIdx.x * 256 + threadIdx.x;
    if (i >= total) return;
    int d = i & 1023;
    int t = (i >> 10) & 1023;
    float xm = (t > 0)    ? x[i - 1024] : 0.0f;
    float x0 = x[i];
    float xp = (t < 1023) ? x[i + 1024] : 0.0f;
    float v = xm * w[d * 3 + 0] + x0 * w[d * 3 + 1] + xp * w[d * 3 + 2] + cb[d];
    v = siluf(v);
    xcf[i] = v;
    xcb[i] = f2bf(v);
}

// ---------------------------------------------------------------------------
// Chunked selective scan.  T=1024 split into NCH=8 chunks of SCH=128 steps.
//   h_t = exp(dt_t*A) . h_{t-1} + dt_t*B_t*x_t ;  y_t = C_t . h_t
// Phase 1: per chunk from h=0, emit chunk product P_c and local final L_c
// Phase 2: h0_{c+1} = P_c . h0_c + L_c  (8-step sequential fix-up)
// Phase 3: rerun chunk scan seeded with h0_c, emit y
// Scratch layout [b][c][n][d] so phase boundaries are d-coalesced.
// ---------------------------------------------------------------------------
#define SCH 128
#define NCH 8

__global__ __launch_bounds__(256) void scan_part1_k(
    const float* __restrict__ dt, const float* __restrict__ xc,
    const float* __restrict__ BC, const float* __restrict__ Adiag,
    float* __restrict__ P, float* __restrict__ Lf) {
    const int T = 1024, D = 1024;
    int d = blockIdx.x * 256 + threadIdx.x;
    int b = blockIdx.y >> 3;
    int c = blockIdx.y & 7;

    float a[16], h[16], p[16];
#pragma unroll
    for (int n = 0; n < 16; ++n) {
        a[n] = Adiag[d * 16 + n];
        h[n] = 0.0f;
        p[n] = 1.0f;
    }

    __shared__ __align__(16) float sBC[64][32];
    float* sflat = &sBC[0][0];
    size_t base   = ((size_t)b * T) * D + d;
    size_t bcbase = (size_t)b * T * 32;

    for (int t0 = c * SCH; t0 < (c + 1) * SCH; t0 += 64) {
        __syncthreads();
#pragma unroll
        for (int j = 0; j < 2; ++j) {
            int ch = threadIdx.x + j * 256;
            cp16_async(&sflat[ch * 4],
                       &BC[bcbase + (size_t)t0 * 32 + ch * 4]);
        }
        WAIT_ASYNC(0);
        __syncthreads();
        for (int tt = 0; tt < 64; ++tt) {
            size_t off = base + (size_t)(t0 + tt) * D;
            float dtv = dt[off];
            float xv  = xc[off];
            float dx  = dtv * xv;
#pragma unroll
            for (int n = 0; n < 16; ++n) {
                float dA = __expf(dtv * a[n]);
                h[n] = dA * h[n] + dx * sBC[tt][n];
                p[n] *= dA;
            }
        }
    }
#pragma unroll
    for (int n = 0; n < 16; ++n) {
        size_t idx = ((size_t)((b * NCH + c) * 16 + n)) * D + d;
        P[idx]  = p[n];
        Lf[idx] = h[n];
    }
}

__global__ __launch_bounds__(256) void scan_part2_k(
    const float* __restrict__ P, const float* __restrict__ Lf,
    float* __restrict__ H0) {
    const int D = 1024;
    int d = blockIdx.x * 256 + threadIdx.x;
    int b = blockIdx.y;
    float h0[16];
#pragma unroll
    for (int n = 0; n < 16; ++n) h0[n] = 0.0f;
    for (int c = 0; c < NCH; ++c) {
#pragma unroll
        for (int n = 0; n < 16; ++n) {
            size_t idx = ((size_t)((b * NCH + c) * 16 + n)) * D + d;
            H0[idx] = h0[n];
            h0[n] = P[idx] * h0[n] + Lf[idx];
        }
    }
}

__global__ __launch_bounds__(256) void scan_part3_k(
    const float* __restrict__ dt, const float* __restrict__ xc,
    const float* __restrict__ BC, const float* __restrict__ Adiag,
    const float* __restrict__ H0, float* __restrict__ y) {
    const int T = 1024, D = 1024;
    int d = blockIdx.x * 256 + threadIdx.x;
    int b = blockIdx.y >> 3;
    int c = blockIdx.y & 7;

    float a[16], h[16];
#pragma unroll
    for (int n = 0; n < 16; ++n) {
        a[n] = Adiag[d * 16 + n];
        h[n] = H0[((size_t)((b * NCH + c) * 16 + n)) * D + d];
    }

    __shared__ __align__(16) float sBC[64][32];
    float* sflat = &sBC[0][0];
    size_t base   = ((size_t)b * T) * D + d;
    size_t bcbase = (size_t)b * T * 32;

    for (int t0 = c * SCH; t0 < (c + 1) * SCH; t0 += 64) {
        __syncthreads();
#pragma unroll
        for (int j = 0; j < 2; ++j) {
            int ch = threadIdx.x + j * 256;
            cp16_async(&sflat[ch * 4],
                       &BC[bcbase + (size_t)t0 * 32 + ch * 4]);
        }
        WAIT_ASYNC(0);
        __syncthreads();
        for (int tt = 0; tt < 64; ++tt) {
            size_t off = base + (size_t)(t0 + tt) * D;
            float dtv = dt[off];
            float xv  = xc[off];
            float dx  = dtv * xv;
            float cv  = 0.0f;
#pragma unroll
            for (int n = 0; n < 16; ++n) {
                float dA = __expf(dtv * a[n]);
                h[n] = dA * h[n] + dx * sBC[tt][n];
                cv = fmaf(h[n], sBC[tt][16 + n], cv);
            }
            y[off] = cv;
        }
    }
}

// ---------------------------------------------------------------------------
// Gate multiply: prod = bf16(y * z)    (z already has SiLU fused in its GEMM)
// ---------------------------------------------------------------------------
__global__ __launch_bounds__(256) void mul_gate_k(const float* __restrict__ y,
                                                  const float* __restrict__ z,
                                                  unsigned short* __restrict__ p,
                                                  int n) {
    int i = blockIdx.x * 256 + threadIdx.x;
    if (i < n) p[i] = f2bf(y[i] * z[i]);
}

// ---------------------------------------------------------------------------
// Launch
// ---------------------------------------------------------------------------
extern "C" void kernel_launch(void* const* d_in, const int* in_sizes, int n_in,
                              void* d_out, int out_size, void* d_ws, size_t ws_size,
                              hipStream_t stream) {
    (void)in_sizes; (void)n_in; (void)out_size; (void)ws_size;
    const int Bsz = 4, T = 1024, D = 1024;
    const int M = Bsz * T;                 // 4096
    const int ND = M * D;                  // 4M elements

    const float* u     = (const float*)d_in[0];
    const float* Win   = (const float*)d_in[1];
    const float* b_in  = (const float*)d_in[2];
    const float* Wg    = (const float*)d_in[3];
    const float* bg    = (const float*)d_in[4];
    const float* Wout  = (const float*)d_in[5];
    const float* bout  = (const float*)d_in[6];
    const float* convw = (const float*)d_in[7];
    const float* convb = (const float*)d_in[8];
    const float* Adiag = (const float*)d_in[9];
    const float* Wb    = (const float*)d_in[10];
    const float* bb    = (const float*)d_in[11];
    const float* Wc    = (const float*)d_in[12];
    const float* bc    = (const float*)d_in[13];
    const float* Wdt   = (const float*)d_in[14];
    const float* bdt   = (const float*)d_in[15];

    char* ws = (char*)d_ws;
    size_t off = 0;
    auto alloc = [&](size_t bytes) -> void* {
        off = (off + 255) & ~(size_t)255;
        void* p = ws + off;
        off += bytes;
        return p;
    };

    unsigned short* ubf    = (unsigned short*)alloc((size_t)ND * 2);
    unsigned short* Winb   = (unsigned short*)alloc((size_t)D * D * 2);
    unsigned short* Wgb    = (unsigned short*)alloc((size_t)D * D * 2);
    unsigned short* Wdtb   = (unsigned short*)alloc((size_t)D * D * 2);
    unsigned short* Woutb  = (unsigned short*)alloc((size_t)D * D * 2);
    unsigned short* Wbcb   = (unsigned short*)alloc((size_t)D * 32 * 2);
    float*          biasBC = (float*)alloc(32 * 4);
    float*          xpre   = (float*)alloc((size_t)ND * 4);
    float*          zbuf   = (float*)alloc((size_t)ND * 4);
    float*          xcf    = (float*)alloc((size_t)ND * 4);
    unsigned short* xcb    = (unsigned short*)alloc((size_t)ND * 2);
    float*          dtb    = (float*)alloc((size_t)ND * 4);
    float*          BCb    = (float*)alloc((size_t)M * 32 * 4);
    float*          yb     = (float*)alloc((size_t)ND * 4);
    unsigned short* prod   = (unsigned short*)alloc((size_t)ND * 2);
    const size_t chElems = (size_t)Bsz * NCH * 16 * D;      // 512K floats
    float*          Pbuf   = (float*)alloc(chElems * 4);
    float*          Lbuf   = (float*)alloc(chElems * 4);
    float*          H0buf  = (float*)alloc(chElems * 4);

    const int DD = D * D;
    cvt_bf16_k<<<(ND + 255) / 256, 256, 0, stream>>>(u, ubf, ND);
    cvt_bf16_k<<<(DD + 255) / 256, 256, 0, stream>>>(Win, Winb, DD);
    cvt_bf16_k<<<(DD + 255) / 256, 256, 0, stream>>>(Wg, Wgb, DD);
    cvt_bf16_k<<<(DD + 255) / 256, 256, 0, stream>>>(Wdt, Wdtb, DD);
    cvt_bf16_k<<<(DD + 255) / 256, 256, 0, stream>>>(Wout, Woutb, DD);
    concat_bc_k<<<(D * 32 + 255) / 256, 256, 0, stream>>>(Wb, Wc, bb, bc, Wbcb,
                                                          biasBC, D);

    dim3 gg(D / BN, M / BM);   // (16, 32)
    // x = u @ Win + b_in
    gemm_bf16_k<0><<<gg, 256, 0, stream>>>(ubf, Winb, b_in, xpre, M, D, D);
    // z = silu(u @ Wg + bg)
    gemm_bf16_k<1><<<gg, 256, 0, stream>>>(ubf, Wgb, bg, zbuf, M, D, D);
    // xc = silu(conv3(x))
    conv_silu_k<<<(ND + 255) / 256, 256, 0, stream>>>(xpre, convw, convb, xcf,
                                                      xcb, ND);
    // dt = softplus(xc @ Wdt + bdt)
    gemm_bf16_k<2><<<gg, 256, 0, stream>>>(xcb, Wdtb, bdt, dtb, M, D, D);
    // [Bm|Cm] = xc @ [Wb|Wc] + [bb|bc]
    gemm_n32_k<<<M / 128, 256, 0, stream>>>(xcb, Wbcb, biasBC, BCb, M, D);
    // chunked selective scan
    dim3 gs(D / 256, Bsz * NCH);            // (4, 32)
    scan_part1_k<<<gs, 256, 0, stream>>>(dtb, xcf, BCb, Adiag, Pbuf, Lbuf);
    scan_part2_k<<<dim3(D / 256, Bsz), 256, 0, stream>>>(Pbuf, Lbuf, H0buf);
    scan_part3_k<<<gs, 256, 0, stream>>>(dtb, xcf, BCb, Adiag, H0buf, yb);
    // gated product -> bf16
    mul_gate_k<<<(ND + 255) / 256, 256, 0, stream>>>(yb, zbuf, prod, ND);
    // out = prod @ Wout + bout
    gemm_bf16_k<0><<<gg, 256, 0, stream>>>(prod, Woutb, bout, (float*)d_out, M,
                                           D, D);
}
